// MultiHeadSelfAttention_12275016532545
// MI455X (gfx1250) — matrix operand order repaired and checked by reading, not compiled, not run on hardware
//
#include <hip/hip_runtime.h>
#include <hip/hip_bf16.h>

// ---------------------------------------------------------------------------
// MHA forward for MI455X (gfx1250, wave32, WMMA).
// B=4, S=2048, D=1024, H=16, dk=dv=64.  All matmuls on v_wmma_f32_16x16x32_bf16
// (fp32 accumulate).  ~137 GFLOP vs ~80MB HBM traffic -> compute bound.
// Data staging uses CDNA5 GLOBAL_LOAD_ASYNC_TO_LDS_B128 (ASYNCcnt) where the
// copy is layout-preserving, with double-buffered LDS tiles in the GEMM so the
// next tile's HBM latency hides under the current tile's WMMAs.
// ---------------------------------------------------------------------------

typedef __attribute__((ext_vector_type(16))) __bf16 v16bf;
typedef __attribute__((ext_vector_type(8)))  float  v8f;

union BF16Frag {
    v16bf v;
    float4 q[2];
    unsigned short u16[16];
};

__device__ __forceinline__ unsigned short f2bf(float f) {
    unsigned int u = __float_as_uint(f);
    unsigned int r = (u + 0x7FFFu + ((u >> 16) & 1u)) >> 16;  // round-nearest-even
    return (unsigned short)r;
}

// CDNA5 async global->LDS 16-byte copy (per-lane addressed, ASYNCcnt-tracked).
// The low 32 bits of a generic pointer derived from __shared__ are the
// workgroup-relative LDS byte address (flat aperture layout), which is what
// the VDST operand of global_load_async_to_lds_* expects.
__device__ __forceinline__ void async_load_b128(const void* gaddr, void* lds) {
    unsigned int l = (unsigned int)(unsigned long long)lds;
    asm volatile("global_load_async_to_lds_b128 %0, %1, off"
                 :: "v"(l), "v"(gaddr) : "memory");
}
__device__ __forceinline__ void wait_async0() {
    asm volatile("s_wait_asynccnt 0" ::: "memory");
}

// ---------------------------------------------------------------------------
// Elementwise fp32 -> bf16 conversion.
// ---------------------------------------------------------------------------
__global__ __launch_bounds__(256) void cvt_f32_bf16_kernel(
    const float* __restrict__ in, unsigned short* __restrict__ out, int n) {
    int i = blockIdx.x * 256 + threadIdx.x;
    if (i < n) out[i] = f2bf(in[i]);
}

// ---------------------------------------------------------------------------
// GEMM: C[M,N] = A[M,K](bf16) * W[N,K](bf16)^T + bias[N]
// Block tile 128x128x32, 256 threads = 8 waves (2 x 4), wave tile 64x32
// = 4x2 accumulators of 16x16.  A staged [M][K] row-major via async-to-LDS,
// W staged [K][N] (transposed scatter).  Double-buffered LDS pipeline.
// outBF16 != 0 -> write bf16 to Cb, else fp32 to Cf.
// ---------------------------------------------------------------------------
__global__ __launch_bounds__(256) void gemm_bf16_wmma_kernel(
    const unsigned short* __restrict__ A,   // [M,K] bf16
    const unsigned short* __restrict__ W,   // [N,K] bf16
    const float* __restrict__ bias,         // [N]
    unsigned short* __restrict__ Cb,        // [M,N] bf16 (if outBF16)
    float* __restrict__ Cf,                 // [M,N] fp32 (if !outBF16)
    int M, int N, int K, int outBF16) {
    __shared__ unsigned short As[2][128][32];  // [buf][m][k]   16 KB
    __shared__ unsigned short Bs[2][32][128];  // [buf][k][n]   16 KB

    const int tid  = threadIdx.x;
    const int lane = tid & 31;
    const int wave = tid >> 5;
    const int wr   = wave >> 2;   // 0..1 : wave row (64 rows each)
    const int wc   = wave & 3;    // 0..3 : wave col (32 cols each)
    const int bm   = blockIdx.y * 128;
    const int bn   = blockIdx.x * 128;

    v8f acc[4][2];
#pragma unroll
    for (int mt = 0; mt < 4; ++mt)
#pragma unroll
        for (int nt = 0; nt < 2; ++nt)
#pragma unroll
            for (int r = 0; r < 8; ++r) acc[mt][nt][r] = 0.0f;

    const int khalf = (lane < 16) ? 0 : 8;                       // A-frag K split
    const int krow  = (lane & 15) + ((lane < 16) ? 0 : 16);      // B-frag K lane

    const int niter = K / 32;

    // ---- prologue: stage tile 0 into buffer 0 ----
#pragma unroll
    for (int i = 0; i < 2; ++i) {
        int id = tid + i * 256;           // 0..511
        int m  = id >> 2;                 // row 0..127
        int c  = id & 3;                  // chunk of 8 bf16
        async_load_b128(&A[(size_t)(bm + m) * K + c * 8], &As[0][m][c * 8]);
        float4 t4 = *(const float4*)&W[(size_t)(bn + (id >> 2)) * K + c * 8];
        const unsigned short* t = (const unsigned short*)&t4;
#pragma unroll
        for (int e = 0; e < 8; ++e) Bs[0][c * 8 + e][id >> 2] = t[e];
    }
    wait_async0();
    __syncthreads();

    for (int kt = 0; kt < niter; ++kt) {
        const int cur = kt & 1;
        const int nxt = cur ^ 1;
        const int k0n = (kt + 1) * 32;
        const bool pre = (kt + 1) < niter;

        // ---- issue next tile's loads before computing on the current one ----
        float4 w4[2];
        if (pre) {
#pragma unroll
            for (int i = 0; i < 2; ++i) {
                int id = tid + i * 256;
                int m  = id >> 2;
                int c  = id & 3;
                async_load_b128(&A[(size_t)(bm + m) * K + k0n + c * 8],
                                &As[nxt][m][c * 8]);
                w4[i] = *(const float4*)&W[(size_t)(bn + m) * K + k0n + c * 8];
            }
        }

        // ---- compute on current buffer: load fragments + 8 WMMAs ----
        BF16Frag a[4], b[2];
#pragma unroll
        for (int mt = 0; mt < 4; ++mt) {
            int m = wr * 64 + mt * 16 + (lane & 15);
            a[mt].q[0] = *(const float4*)&As[cur][m][khalf];
            a[mt].q[1] = *(const float4*)&As[cur][m][khalf + 16];
        }
#pragma unroll
        for (int nt = 0; nt < 2; ++nt) {
            int n = wc * 32 + nt * 16 + (lane & 15);
#pragma unroll
            for (int e = 0; e < 8; ++e) {
                b[nt].u16[e]     = Bs[cur][khalf + e][n];
                b[nt].u16[8 + e] = Bs[cur][khalf + 16 + e][n];
            }
        }
#pragma unroll
        for (int mt = 0; mt < 4; ++mt)
#pragma unroll
            for (int nt = 0; nt < 2; ++nt)
                acc[mt][nt] = __builtin_amdgcn_wmma_f32_16x16x32_bf16(
                    false, a[mt].v, false, b[nt].v, (short)0, acc[mt][nt],
                    false, false);

        // ---- finish staging the next tile, then rendezvous ----
        if (pre) {
#pragma unroll
            for (int i = 0; i < 2; ++i) {
                int id = tid + i * 256;
                int n  = id >> 2;
                int c  = id & 3;
                const unsigned short* t = (const unsigned short*)&w4[i];
#pragma unroll
                for (int e = 0; e < 8; ++e) Bs[nxt][c * 8 + e][n] = t[e];
            }
            wait_async0();
        }
        __syncthreads();
    }

    // ---- epilogue: bias + store ----
#pragma unroll
    for (int mt = 0; mt < 4; ++mt)
#pragma unroll
        for (int nt = 0; nt < 2; ++nt)
#pragma unroll
            for (int r = 0; r < 8; ++r) {
                int row = bm + wr * 64 + mt * 16 + r + ((lane < 16) ? 0 : 8);
                int col = bn + wc * 32 + nt * 16 + (lane & 15);
                float v = acc[mt][nt][r] + bias[col];
                if (outBF16) Cb[(size_t)row * N + col] = f2bf(v);
                else         Cf[(size_t)row * N + col] = v;
            }
}

// ---------------------------------------------------------------------------
// Flash attention. qkv: [B*S, 3072] bf16 (q | k | v, head-major 64 each).
// O: [B*S, 1024] bf16 (head-interleaved, ready for the projection GEMM).
// Grid: (S/128, H, B).  256 threads = 8 waves; wave w owns q rows
// [128*bx + 16*w, +16).  Key loop steps 32 keys; block stages K^T (manual
// transpose) and V (async-to-LDS) each step.
// ---------------------------------------------------------------------------
__global__ __launch_bounds__(256) void flash_attn_wmma_kernel(
    const unsigned short* __restrict__ qkv, unsigned short* __restrict__ O) {
    constexpr int S = 2048, QKVW = 3072, OW = 1024, HD = 64;
    __shared__ unsigned short Kt[64][32];      // [dk][key]   4 KB
    __shared__ unsigned short Vt[32][64];      // [key][dv]   4 KB
    __shared__ unsigned short Pw[8][16][32];   // per-wave P  8 KB

    const int b = blockIdx.z, h = blockIdx.y;
    const int qbase = blockIdx.x * 128;
    const int tid = threadIdx.x, lane = tid & 31, wave = tid >> 5;

    const size_t base = (size_t)b * S * QKVW;
    const int qcol = h * HD;
    const int kcol = 1024 + h * HD;
    const int vcol = 2048 + h * HD;

    const int khalf = (lane < 16) ? 0 : 8;
    const int krow  = (lane & 15) + ((lane < 16) ? 0 : 16);

    // ---- load Q fragments (16 x 64 per wave, two K=32 chunks) ----
    BF16Frag aq[2];
    {
        int m = qbase + wave * 16 + (lane & 15);
        const unsigned short* qrow = qkv + base + (size_t)m * QKVW + qcol;
#pragma unroll
        for (int kc = 0; kc < 2; ++kc) {
            aq[kc].q[0] = *(const float4*)&qrow[kc * 32 + khalf];
            aq[kc].q[1] = *(const float4*)&qrow[kc * 32 + khalf + 16];
        }
    }

    float mrun[8], lrun[8];
    v8f o[4];
#pragma unroll
    for (int r = 0; r < 8; ++r) { mrun[r] = -1e30f; lrun[r] = 0.0f; }
#pragma unroll
    for (int nt = 0; nt < 4; ++nt)
#pragma unroll
        for (int r = 0; r < 8; ++r) o[nt][r] = 0.0f;

    const float SC = 0.125f;  // 64^-0.5

    for (int j = 0; j < S; j += 32) {
        __syncthreads();
        // ---- stage V (async-to-LDS) and K^T (transposed scatter) ----
        {
            int n = tid >> 3;   // key 0..31
            int c = tid & 7;    // dk chunk of 8
            async_load_b128(&qkv[base + (size_t)(j + n) * QKVW + vcol + c * 8],
                            &Vt[n][c * 8]);
            float4 t4 =
                *(const float4*)&qkv[base + (size_t)(j + n) * QKVW + kcol + c * 8];
            const unsigned short* t = (const unsigned short*)&t4;
#pragma unroll
            for (int e = 0; e < 8; ++e) Kt[c * 8 + e][n] = t[e];
        }
        wait_async0();
        __syncthreads();

        // ---- scores S = Q * K^T (16 x 32), 4 WMMAs ----
        v8f s[2];
#pragma unroll
        for (int nt = 0; nt < 2; ++nt)
#pragma unroll
            for (int r = 0; r < 8; ++r) s[nt][r] = 0.0f;
#pragma unroll
        for (int kc = 0; kc < 2; ++kc)
#pragma unroll
            for (int nt = 0; nt < 2; ++nt) {
                BF16Frag bk;
#pragma unroll
                for (int e = 0; e < 8; ++e) {
                    bk.u16[e]     = Kt[kc * 32 + khalf + e][nt * 16 + (lane & 15)];
                    bk.u16[8 + e] = Kt[kc * 32 + khalf + 16 + e][nt * 16 + (lane & 15)];
                }
                s[nt] = __builtin_amdgcn_wmma_f32_16x16x32_bf16(
                    false, aq[kc].v, false, bk.v, (short)0, s[nt], false, false);
            }

        // ---- online softmax (rows live per-VGPR, across 16-lane halves) ----
#pragma unroll
        for (int r = 0; r < 8; ++r) {
            float s0 = s[0][r] * SC, s1 = s[1][r] * SC;
            float mx = fmaxf(s0, s1);
#pragma unroll
            for (int off = 8; off >= 1; off >>= 1)
                mx = fmaxf(mx, __shfl_xor(mx, off, 32));
            float mnew = fmaxf(mrun[r], mx);
            float fac  = __expf(mrun[r] - mnew);
            float p0 = __expf(s0 - mnew);
            float p1 = __expf(s1 - mnew);
            float rs = p0 + p1;
#pragma unroll
            for (int off = 8; off >= 1; off >>= 1)
                rs += __shfl_xor(rs, off, 32);
            lrun[r] = lrun[r] * fac + rs;
            mrun[r] = mnew;
#pragma unroll
            for (int nt = 0; nt < 4; ++nt) o[nt][r] *= fac;
            int prow = r + ((lane < 16) ? 0 : 8);
            Pw[wave][prow][lane & 15]        = f2bf(p0);
            Pw[wave][prow][16 + (lane & 15)] = f2bf(p1);
        }
        // wave-private LDS round-trip: keep the P stores ahead of the reloads
        asm volatile("s_wait_dscnt 0" ::: "memory");

        // ---- reload P as an A-fragment (16 x 32) ----
        BF16Frag ap;
        {
            int pm = lane & 15;
            ap.q[0] = *(const float4*)&Pw[wave][pm][khalf];
            ap.q[1] = *(const float4*)&Pw[wave][pm][khalf + 16];
        }
        // ---- O += P * V, 4 WMMAs ----
#pragma unroll
        for (int nt = 0; nt < 4; ++nt) {
            BF16Frag bv;
#pragma unroll
            for (int e = 0; e < 8; ++e) {
                bv.u16[e]     = Vt[khalf + e][nt * 16 + (lane & 15)];
                bv.u16[8 + e] = Vt[khalf + 16 + e][nt * 16 + (lane & 15)];
            }
            o[nt] = __builtin_amdgcn_wmma_f32_16x16x32_bf16(
                false, ap.v, false, bv.v, (short)0, o[nt], false, false);
        }
    }

    // ---- epilogue: normalize, write bf16 [B*S, H*64] ----
#pragma unroll
    for (int nt = 0; nt < 4; ++nt)
#pragma unroll
        for (int r = 0; r < 8; ++r) {
            int row = qbase + wave * 16 + r + ((lane < 16) ? 0 : 8);
            int col = h * HD + nt * 16 + (lane & 15);
            float v = o[nt][r] / lrun[r];
            O[(size_t)(b * S + row) * OW + col] = f2bf(v);
        }
}

// ---------------------------------------------------------------------------
// Host-side launcher.
// ---------------------------------------------------------------------------
extern "C" void kernel_launch(void* const* d_in, const int* in_sizes, int n_in,
                              void* d_out, int out_size, void* d_ws, size_t ws_size,
                              hipStream_t stream) {
    (void)in_sizes; (void)n_in; (void)out_size; (void)ws_size;
    constexpr int B = 4, S = 2048, D = 1024;
    constexpr int M = B * S;               // 8192
    constexpr int NQKV = 3072;

    const float* x      = (const float*)d_in[0];  // [B,S,D]
    const float* qkv_w  = (const float*)d_in[1];  // [3072, D]
    const float* qkv_b  = (const float*)d_in[2];  // [3072]
    const float* proj_w = (const float*)d_in[3];  // [D, 1024]
    const float* proj_b = (const float*)d_in[4];  // [D]
    float*       out    = (float*)d_out;          // [B,S,D] fp32

    // workspace layout (bytes, all 256-aligned)
    char* ws = (char*)d_ws;
    unsigned short* xb    = (unsigned short*)(ws);                    // 16 MB
    unsigned short* wqkvb = (unsigned short*)(ws + (size_t)16777216); //  6 MB
    unsigned short* qkvb  = (unsigned short*)(ws + (size_t)23068672); // 48 MB
    unsigned short* ob    = (unsigned short*)(ws + (size_t)73400320); // 16 MB
    unsigned short* wprjb = (unsigned short*)(ws + (size_t)90177536); //  2 MB

    // 1) convert activations + weights to bf16
    {
        int n = M * D;
        cvt_f32_bf16_kernel<<<(n + 255) / 256, 256, 0, stream>>>(x, xb, n);
        n = NQKV * D;
        cvt_f32_bf16_kernel<<<(n + 255) / 256, 256, 0, stream>>>(qkv_w, wqkvb, n);
        n = D * D;
        cvt_f32_bf16_kernel<<<(n + 255) / 256, 256, 0, stream>>>(proj_w, wprjb, n);
    }

    // 2) QKV GEMM: qkv[M,3072] = xb * wqkvb^T + qkv_b   (bf16 out)
    {
        dim3 grid(NQKV / 128, M / 128);
        gemm_bf16_wmma_kernel<<<grid, 256, 0, stream>>>(
            xb, wqkvb, qkv_b, qkvb, nullptr, M, NQKV, D, 1);
    }

    // 3) flash attention -> ob[M,1024] bf16
    {
        dim3 grid(S / 128, 16 /*H*/, B);
        flash_attn_wmma_kernel<<<grid, 256, 0, stream>>>(qkvb, ob);
    }

    // 4) projection GEMM: out[M,D] = ob * wprjb^T + proj_b   (fp32 out)
    {
        dim3 grid(D / 128, M / 128);
        gemm_bf16_wmma_kernel<<<grid, 256, 0, stream>>>(
            ob, wprjb, proj_b, nullptr, out, M, D, D, 0);
    }
}
